// GATv2Encoder_1159641170555
// MI455X (gfx1250) — matrix operand order repaired and checked by reading, not compiled, not run on hardware
//
#include <hip/hip_runtime.h>
#include <hip/hip_bf16.h>
#include <math.h>

typedef __attribute__((ext_vector_type(16))) __bf16 v16bf;
typedef __attribute__((ext_vector_type(8)))  float  v8f;

#define NEG_SLOPE 0.2f
#define BN_EPS 1e-5f

// ---------------------------------------------------------------- utilities

__global__ void k_fill_f32(float* __restrict__ p, float v, long n) {
    long i = (long)blockIdx.x * blockDim.x + threadIdx.x;
    if (i < n) p[i] = v;
}

// out[n, H] = bias[H] broadcast (segment-sum accumulator init)
__global__ void k_init_bias(float* __restrict__ out, const float* __restrict__ bias,
                            int N, int H) {
    long i = (long)blockIdx.x * blockDim.x + threadIdx.x;
    if (i >= (long)N * H) return;
    out[i] = bias[i % H];
}

// f32 [rows, cols] -> bf16 [rows, kpad], zero padded
__global__ void k_cast_pad(const float* __restrict__ src, __bf16* __restrict__ dst,
                           int rows, int cols, int kpad) {
    long i = (long)blockIdx.x * blockDim.x + threadIdx.x;
    if (i >= (long)rows * kpad) return;
    int k = (int)(i % kpad);
    long r = i / kpad;
    float v = (k < cols) ? src[r * cols + k] : 0.0f;
    dst[i] = (__bf16)v;
}

// weight f32 [K, H] -> bf16 [H, kpad] (transposed, zero padded)
__global__ void k_cast_transpose(const float* __restrict__ src, __bf16* __restrict__ dst,
                                 int K, int H, int kpad) {
    int i = blockIdx.x * blockDim.x + threadIdx.x;
    if (i >= H * kpad) return;
    int h = i / kpad, k = i % kpad;
    dst[i] = (__bf16)((k < K) ? src[k * H + h] : 0.0f);
}

// ---------------------------------------------------------------- WMMA GEMM (dual output)
// C0[M,H] = A[M,Kpad] x B0,  C1[M,H] = A x B1  (B given transposed: BT[H,Kpad], bf16)
// Block = 8 waves (32x8). Each wave owns one 16-wide column tile and computes BOTH
// outputs, sharing the A-operand registers across two v_wmma accumulator chains.
// Layouts used here (contiguous k order, not the ISA table order):
//  A 16x32 bf16 : lane L holds row M=L%16; elem e -> K=k0+16*(L/16)+e
//  B 32x16 bf16 : lane L holds col N=L%16; elem e -> K=k0+16*(L/16)+e
//  C 16x16 f32  : lane L col N=L%16; vgpr r -> row M=r+8*(L/16)
__global__ void k_wmma_gemm_dual(const __bf16* __restrict__ A,
                                 const __bf16* __restrict__ BT0,
                                 const __bf16* __restrict__ BT1,
                                 float* __restrict__ C0, float* __restrict__ C1,
                                 int M, int Hc, int Kpad) {
    const int lane = threadIdx.x;                       // 0..31
    const int ct   = blockIdx.y * blockDim.y + threadIdx.y;  // column tile
    if (ct * 16 >= Hc) return;                          // whole wave exits together

    const int half = lane >> 4;                         // 0 or 1
    const int l16  = lane & 15;
    const int row  = blockIdx.x * 16 + l16;
    const int col  = ct * 16 + l16;

    const __bf16* __restrict__ Arow = A   + (long)row * Kpad;
    const __bf16* __restrict__ B0   = BT0 + (long)col * Kpad;
    const __bf16* __restrict__ B1   = BT1 + (long)col * Kpad;

    v8f acc0 = {}, acc1 = {};
    const int abase = half * 8;
    const int bbase = half * 16;

    for (int k0 = 0; k0 < Kpad; k0 += 32) {
        v16bf a, b0, b1;
        const __bf16* a0 = Arow + k0 + 2 * abase;
        const __bf16* a1 = Arow + k0 + 8 + 2 * abase;
        const __bf16* p0 = B0 + k0 + bbase;
        const __bf16* p1 = B1 + k0 + bbase;
#pragma unroll
        for (int e = 0; e < 8; ++e) { a[e] = a0[e]; a[8 + e] = a1[e]; }
#pragma unroll
        for (int e = 0; e < 16; ++e) { b0[e] = p0[e]; b1[e] = p1[e]; }
        acc0 = __builtin_amdgcn_wmma_f32_16x16x32_bf16(
            false, a, false, b0, (short)0, acc0, false, false);
        acc1 = __builtin_amdgcn_wmma_f32_16x16x32_bf16(
            false, a, false, b1, (short)0, acc1, false, false);
    }

    const long base = (long)(blockIdx.x * 16) * Hc + ct * 16;
#pragma unroll
    for (int r = 0; r < 8; ++r) {
        long off = base + (long)(r + half * 8) * Hc + l16;
        C0[off] = acc0[r];
        C1[off] = acc1[r];
    }
}

// ---------------------------------------------------------------- edge phase

__device__ __forceinline__ void get_edge(const int* __restrict__ ei, int E0, int e,
                                         int& s, int& d) {
    if (e < E0) { s = ei[e]; d = ei[E0 + e]; }
    else        { s = e - E0; d = e - E0; }   // self loops appended
}

__device__ __forceinline__ void atomicMaxF(float* addr, float v) {
    if (v >= 0.0f) atomicMax((int*)addr, __float_as_int(v));
    else           atomicMin((unsigned int*)addr, (unsigned int)__float_as_int(v));
}

// logits[e,h] = att[h] . leaky_relu(xl[src] + xr[dst]);  m[dst,h] = max(...)
__global__ void k_edge_logits_max(const float* __restrict__ xl, const float* __restrict__ xr,
                                  const float* __restrict__ att, const int* __restrict__ ei,
                                  int E0, int ET, int heads, int C,
                                  float* __restrict__ logits, float* __restrict__ mmax) {
    long i = (long)blockIdx.x * blockDim.x + threadIdx.x;
    if (i >= (long)ET * heads) return;
    int e = (int)(i / heads), h = (int)(i % heads);
    int s, d; get_edge(ei, E0, e, s, d);
    const float4* pl = reinterpret_cast<const float4*>(xl + (long)s * heads * C + h * C);
    const float4* pr = reinterpret_cast<const float4*>(xr + (long)d * heads * C + h * C);
    const float4* pa = reinterpret_cast<const float4*>(att + h * C);
    float acc = 0.0f;
    for (int c = 0; c < C / 4; ++c) {
        float4 l = pl[c], r = pr[c], a = pa[c];
        float v;
        v = l.x + r.x; acc += a.x * ((v > 0.0f) ? v : v * NEG_SLOPE);
        v = l.y + r.y; acc += a.y * ((v > 0.0f) ? v : v * NEG_SLOPE);
        v = l.z + r.z; acc += a.z * ((v > 0.0f) ? v : v * NEG_SLOPE);
        v = l.w + r.w; acc += a.w * ((v > 0.0f) ? v : v * NEG_SLOPE);
    }
    logits[i] = acc;
    atomicMaxF(&mmax[(long)d * heads + h], acc);
}

// logits[e,h] <- exp(logit - m[dst,h]);  denom[dst,h] += exp
__global__ void k_edge_exp_sum(const float* __restrict__ mmax, const int* __restrict__ ei,
                               int E0, int ET, int heads,
                               float* __restrict__ logits, float* __restrict__ denom) {
    long i = (long)blockIdx.x * blockDim.x + threadIdx.x;
    if (i >= (long)ET * heads) return;
    int e = (int)(i / heads), h = (int)(i % heads);
    int s, d; get_edge(ei, E0, e, s, d);
    float ex = expf(logits[i] - mmax[(long)d * heads + h]);
    logits[i] = ex;
    atomicAdd(&denom[(long)d * heads + h], ex);
}

// one thread per (edge, head*C+c): fully coalesced gather + atomic scatter
__global__ void k_edge_aggregate(const float* __restrict__ xl, const float* __restrict__ logits,
                                 const float* __restrict__ denom, const int* __restrict__ ei,
                                 int E0, int ET, int heads, int C,
                                 float* __restrict__ out) {
    long i = (long)blockIdx.x * blockDim.x + threadIdx.x;
    const int HC = heads * C;
    if (i >= (long)ET * HC) return;
    int hc = (int)(i % HC);
    int e  = (int)(i / HC);
    int h  = hc / C;
    int s, d; get_edge(ei, E0, e, s, d);
    float alpha = logits[(long)e * heads + h] / (denom[(long)d * heads + h] + 1e-16f);
    atomicAdd(&out[(long)d * HC + hc], xl[(long)s * HC + hc] * alpha);
}

// ---------------------------------------------------------------- batchnorm + ELU

// one block per channel: mu and rsqrt(var+eps)
__global__ void k_bn_stats(const float* __restrict__ h, int N, int H,
                           float* __restrict__ stats /* [2*H]: mu, rs */) {
    int ch = blockIdx.x;
    float s = 0.0f, ss = 0.0f;
    for (int n = threadIdx.x; n < N; n += blockDim.x) {
        float v = h[(long)n * H + ch];
        s += v; ss += v * v;
    }
    __shared__ float sh[256], sh2[256];
    sh[threadIdx.x] = s; sh2[threadIdx.x] = ss;
    __syncthreads();
    for (int st = blockDim.x >> 1; st > 0; st >>= 1) {
        if (threadIdx.x < st) {
            sh[threadIdx.x]  += sh[threadIdx.x + st];
            sh2[threadIdx.x] += sh2[threadIdx.x + st];
        }
        __syncthreads();
    }
    if (threadIdx.x == 0) {
        float mu  = sh[0] / (float)N;
        float var = sh2[0] / (float)N - mu * mu;
        stats[ch]     = mu;
        stats[H + ch] = rsqrtf(var + BN_EPS);
    }
}

__global__ void k_bn_elu(const float* __restrict__ h, const float* __restrict__ stats,
                         const float* __restrict__ gamma, const float* __restrict__ beta,
                         float* __restrict__ out, int N, int H) {
    long i = (long)blockIdx.x * blockDim.x + threadIdx.x;
    if (i >= (long)N * H) return;
    int ch = (int)(i % H);
    float v = (h[i] - stats[ch]) * stats[H + ch] * gamma[ch] + beta[ch];
    out[i] = (v > 0.0f) ? v : (expf(v) - 1.0f);
}

// ---------------------------------------------------------------- graph pooling

__global__ void k_pool_sum(const float* __restrict__ emb, const int* __restrict__ batch,
                           int N, int C, float* __restrict__ sums, float* __restrict__ counts) {
    long i = (long)blockIdx.x * blockDim.x + threadIdx.x;
    if (i >= (long)N * C) return;
    int n = (int)(i / C), c = (int)(i % C);
    int g = batch[n];
    atomicAdd(&sums[(long)g * C + c], emb[i]);
    if (c == 0) atomicAdd(&counts[g], 1.0f);
}

__global__ void k_pool_div(const float* __restrict__ sums, const float* __restrict__ counts,
                           float* __restrict__ out, int G, int C) {
    int i = blockIdx.x * blockDim.x + threadIdx.x;
    if (i >= G * C) return;
    out[i] = sums[i] / fmaxf(counts[i / C], 1.0f);
}

// ---------------------------------------------------------------- driver

static inline int cdiv(long a, long b) { return (int)((a + b - 1) / b); }

extern "C" void kernel_launch(void* const* d_in, const int* in_sizes, int n_in,
                              void* d_out, int out_size, void* d_ws, size_t ws_size,
                              hipStream_t stream) {
    const int N      = in_sizes[2];          // 50000 (batch length)
    const int E0     = in_sizes[1] / 2;      // 1,600,000
    const int ET     = E0 + N;               // + self loops
    const int IN_CH  = 14, HID = 32, HEADS = 4, H = HID * HEADS; // 128
    const int NG     = 512;

    const float* x   = (const float*)d_in[0];
    const int* ei    = (const int*)d_in[1];
    const int* batch = (const int*)d_in[2];
    const float* Wl[3]  = { (const float*)d_in[3],  (const float*)d_in[9],  (const float*)d_in[15] };
    const float* Wr[3]  = { (const float*)d_in[4],  (const float*)d_in[10], (const float*)d_in[16] };
    const float* att[3] = { (const float*)d_in[5],  (const float*)d_in[11], (const float*)d_in[17] };
    const float* bi[3]  = { (const float*)d_in[6],  (const float*)d_in[12], (const float*)d_in[18] };
    const float* ga[3]  = { (const float*)d_in[7],  (const float*)d_in[13], (const float*)d_in[19] };
    const float* be[3]  = { (const float*)d_in[8],  (const float*)d_in[14], (const float*)d_in[20] };

    // ---- carve workspace
    char* p = (char*)d_ws;
    auto carve = [&](size_t bytes) { void* r = p; p += (bytes + 255) & ~size_t(255); return r; };
    float*  hbuf   = (float*) carve((size_t)N * H * 4);        // activations (in/out per layer)
    float*  xl     = (float*) carve((size_t)N * H * 4);
    float*  xr     = (float*) carve((size_t)N * H * 4);
    __bf16* Abf    = (__bf16*)carve((size_t)N * H * 2);        // bf16 GEMM A
    __bf16* WlT    = (__bf16*)carve((size_t)H * H * 2);
    __bf16* WrT    = (__bf16*)carve((size_t)H * H * 2);
    float*  logits = (float*) carve((size_t)ET * HEADS * 4);
    float*  mmax   = (float*) carve((size_t)N * HEADS * 4);
    float*  denom  = (float*) carve((size_t)N * HEADS * 4);
    float*  stats  = (float*) carve((size_t)2 * H * 4);
    float*  psum   = (float*) carve((size_t)NG * HID * 4);
    float*  pcnt   = (float*) carve((size_t)NG * 4);
    (void)ws_size; (void)n_in; (void)out_size;

    float* graph_emb = (float*)d_out;                  // [512, 32]
    float* node_emb  = (float*)d_out + NG * HID;       // [50000, 32]

    const int TB = 256;

    for (int L = 0; L < 3; ++L) {
        const int Kin   = (L == 0) ? IN_CH : H;          // 14, 128, 128
        const int Kpad  = (L == 0) ? 32 : 128;
        const int heads = (L == 2) ? 1 : HEADS;
        const int C     = HID;
        const int Hout  = heads * C;                     // 128, 128, 32
        const float* hin = (L == 0) ? x : hbuf;

        // bf16 cast/pad of activations and transposed weights
        k_cast_pad<<<cdiv((long)N * Kpad, TB), TB, 0, stream>>>(hin, Abf, N, Kin, Kpad);
        k_cast_transpose<<<cdiv(Hout * Kpad, TB), TB, 0, stream>>>(Wl[L], WlT, Kin, Hout, Kpad);
        k_cast_transpose<<<cdiv(Hout * Kpad, TB), TB, 0, stream>>>(Wr[L], WrT, Kin, Hout, Kpad);

        // xl = h @ Wl and xr = h @ Wr fused: shared A regs, two WMMA chains per wave
        {
            dim3 blk(32, 8);
            dim3 grd(N / 16, cdiv(Hout / 16, 8));
            k_wmma_gemm_dual<<<grd, blk, 0, stream>>>(Abf, WlT, WrT, xl, xr, N, Hout, Kpad);
        }

        // attention softmax over incoming edges per destination node
        long nh = (long)N * heads, eh = (long)ET * heads, ehc = (long)ET * heads * C;
        k_fill_f32<<<cdiv(nh, TB), TB, 0, stream>>>(mmax, -INFINITY, nh);
        k_fill_f32<<<cdiv(nh, TB), TB, 0, stream>>>(denom, 0.0f, nh);
        k_init_bias<<<cdiv((long)N * Hout, TB), TB, 0, stream>>>(hbuf, bi[L], N, Hout);

        k_edge_logits_max<<<cdiv(eh, TB), TB, 0, stream>>>(xl, xr, att[L], ei, E0, ET, heads, C, logits, mmax);
        k_edge_exp_sum   <<<cdiv(eh, TB), TB, 0, stream>>>(mmax, ei, E0, ET, heads, logits, denom);
        k_edge_aggregate <<<cdiv(ehc, TB), TB, 0, stream>>>(xl, logits, denom, ei, E0, ET, heads, C, hbuf);

        // batchnorm + ELU (layer 3 writes node embeddings straight into d_out)
        float* bn_out = (L == 2) ? node_emb : hbuf;
        k_bn_stats<<<Hout, 256, 0, stream>>>(hbuf, N, Hout, stats);
        k_bn_elu<<<cdiv((long)N * Hout, TB), TB, 0, stream>>>(hbuf, stats, ga[L], be[L], bn_out, N, Hout);
    }

    // global mean pooling per graph
    k_fill_f32<<<cdiv((long)NG * HID, TB), TB, 0, stream>>>(psum, 0.0f, (long)NG * HID);
    k_fill_f32<<<cdiv((long)NG, TB), TB, 0, stream>>>(pcnt, 0.0f, NG);
    k_pool_sum<<<cdiv((long)N * HID, TB), TB, 0, stream>>>(node_emb, batch, N, HID, psum, pcnt);
    k_pool_div<<<cdiv(NG * HID, TB), TB, 0, stream>>>(psum, pcnt, graph_emb, NG, HID);
}
